// DirectHead_35673998360868
// MI455X (gfx1250) — matrix operand order repaired and checked by reading, not compiled, not run on hardware
//
#include <hip/hip_runtime.h>

typedef __attribute__((ext_vector_type(16))) __bf16 v16bf;
typedef __attribute__((ext_vector_type(8)))  __bf16 v8bf;
typedef __attribute__((ext_vector_type(8)))  float  v8f;
typedef __attribute__((ext_vector_type(4)))  int    v4i;

// ---- CDNA5 async global->LDS copy (ASYNCcnt path), with safe fallback ----
#if defined(__AMDGCN__) && __has_builtin(__builtin_amdgcn_global_load_async_to_lds_b128)
#define DH_ASYNC 1
#else
#define DH_ASYNC 0
#endif

__device__ __forceinline__ void async_g2l_b128(const void* g, void* l) {
#if DH_ASYNC
  __builtin_amdgcn_global_load_async_to_lds_b128(
      (__attribute__((address_space(1))) v4i*)g,
      (__attribute__((address_space(3))) v4i*)l, 0, 0);
#else
  *(v8bf*)l = *(const v8bf*)g;
#endif
}
__device__ __forceinline__ void async_wait0() {
#if DH_ASYNC
#if __has_builtin(__builtin_amdgcn_s_wait_asynccnt)
  __builtin_amdgcn_s_wait_asynccnt(0);
#else
  asm volatile("s_wait_asynccnt 0x0" ::: "memory");
#endif
#endif
}

__device__ __forceinline__ v8f v8f_zero() {
  v8f z;
#pragma unroll
  for (int k = 0; k < 8; ++k) z[k] = 0.f;
  return z;
}
__device__ __forceinline__ v8bf v8bf_zero() {
  v8bf z;
#pragma unroll
  for (int k = 0; k < 8; ++k) z[k] = (__bf16)0.f;
  return z;
}

__device__ __forceinline__ v8f wmma_bf16(v16bf a, v16bf b, v8f c) {
  return __builtin_amdgcn_wmma_f32_16x16x32_bf16(false, a, false, b, (short)0, c, false, false);
}

// A fragment: LDS row-major [rows][stride], tile (m0, k0). Per ISA 16-bit A layout:
// lanes 0-15 : row m0+lane, K {k0..k0+7, k0+16..k0+23}
// lanes 16-31: row m0+lane-16, K {k0+8..k0+15, k0+24..k0+31}
__device__ __forceinline__ v16bf frag_a(const __bf16* lds, int stride, int m0, int k0, int lane) {
  const int lh = (lane >> 4) & 1;
  const __bf16* p = lds + (m0 + (lane & 15)) * stride + k0 + lh * 8;
  v8bf lo = *(const v8bf*)p;
  v8bf hi = *(const v8bf*)(p + 16);
  return __builtin_shufflevector(lo, hi, 0,1,2,3,4,5,6,7,8,9,10,11,12,13,14,15);
}
// B fragment: LDS row-major [N][K], tile (n0, k0). Per ISA 16-bit B layout:
// lanes 0-15 : col n0+lane, K {k0..k0+7, k0+16..k0+23} ; lanes 16-31: col n0+lane-16, K {k0+8..k0+15, k0+24..k0+31}
__device__ __forceinline__ v16bf frag_b(const __bf16* lds, int stride, int n0, int k0, int lane) {
  const int lh = (lane >> 4) & 1;
  const __bf16* p = lds + (n0 + (lane & 15)) * stride + k0 + lh * 8;
  v8bf lo = *(const v8bf*)p;
  v8bf hi = *(const v8bf*)(p + 16);
  return __builtin_shufflevector(lo, hi, 0,1,2,3,4,5,6,7,8,9,10,11,12,13,14,15);
}

// ---------------- elementwise helpers ----------------
__global__ void f32_to_bf16_kernel(const float* __restrict__ s, __bf16* __restrict__ d, int n) {
  int i = blockIdx.x * 256 + threadIdx.x;
  if (i < n) d[i] = (__bf16)s[i];
}
__global__ void split_visual_kernel(const float* __restrict__ vf, __bf16* __restrict__ cls,
                                    __bf16* __restrict__ patch, int n) {
  int i = blockIdx.x * 256 + threadIdx.x;
  if (i >= n) return;
  int d = i % 768; int rb = i / 768; int r = rb % 197; int b = rb / 197;
  __bf16 v = (__bf16)vf[i];
  if (r == 0) cls[(size_t)b * 768 + d] = v;
  else        patch[((size_t)b * 196 + (r - 1)) * 768 + d] = v;
}
__global__ void zero_f32_kernel(float* p, int n) {
  int i = blockIdx.x * 256 + threadIdx.x;
  if (i < n) p[i] = 0.f;
}
__global__ void inv_norm_kernel(const float* __restrict__ sq, float* __restrict__ inv, int n) {
  int i = blockIdx.x * 256 + threadIdx.x;
  if (i < n) inv[i] = 1.f / fmaxf(sqrtf(fmaxf(sq[i], 0.f)), 1e-12f);
}
__global__ void scale_rows_bf16_kernel(const float* __restrict__ x, const float* __restrict__ inv,
                                       __bf16* __restrict__ y, int n, int cols) {
  int i = blockIdx.x * 256 + threadIdx.x;
  if (i < n) y[i] = (__bf16)(x[i] * inv[i / cols]);
}

// ---------------- GEMM: C[M][N] = A[M][K](bf16) * B[N][K]^T(bf16) + bias ----------------
// Optionally writes fp32 C, bf16 C, and accumulates per-row sum-of-squares (for l2norm).
__global__ __launch_bounds__(256) void gemm_bf16_kernel(
    const __bf16* __restrict__ A, const __bf16* __restrict__ B, const float* __restrict__ bias,
    float* Cf, __bf16* Cb, float* rowsq, int M, int N, int K) {
  __shared__ __align__(16) __bf16 sA[128 * 40];
  __shared__ __align__(16) __bf16 sB[128 * 40];
  const int tid = threadIdx.x, lane = tid & 31, wid = tid >> 5;
  const int wm = wid >> 1, wn = wid & 1;
  const int bm = blockIdx.x * 128, bn = blockIdx.y * 128;
  v8f acc[2][4];
#pragma unroll
  for (int m = 0; m < 2; ++m)
#pragma unroll
    for (int n = 0; n < 4; ++n) acc[m][n] = v8f_zero();

  for (int k0 = 0; k0 < K; k0 += 32) {
    __syncthreads();
#pragma unroll
    for (int c0 = 0; c0 < 2; ++c0) {
      int c = tid + c0 * 256;
      int r = c >> 2, kk = (c & 3) * 8;
      async_g2l_b128(A + (size_t)(bm + r) * K + k0 + kk, sA + r * 40 + kk);
      async_g2l_b128(B + (size_t)(bn + r) * K + k0 + kk, sB + r * 40 + kk);
    }
    async_wait0();
    __syncthreads();
    v16bf a0 = frag_a(sA, 40, wm * 32, 0, lane);
    v16bf a1 = frag_a(sA, 40, wm * 32 + 16, 0, lane);
#pragma unroll
    for (int n = 0; n < 4; ++n) {
      v16bf b = frag_b(sB, 40, wn * 64 + n * 16, 0, lane);
      acc[0][n] = wmma_bf16(a0, b, acc[0][n]);
      acc[1][n] = wmma_bf16(a1, b, acc[1][n]);
    }
  }
  const int lh = lane >> 4, c16 = lane & 15;
#pragma unroll
  for (int mi = 0; mi < 2; ++mi) {
#pragma unroll
    for (int r = 0; r < 8; ++r) {
      int row = bm + wm * 32 + mi * 16 + lh * 8 + r;
      float part = 0.f;
#pragma unroll
      for (int n = 0; n < 4; ++n) {
        int col = bn + wn * 64 + n * 16 + c16;
        float v = acc[mi][n][r] + bias[col];
        if (Cf) Cf[(size_t)row * N + col] = v;
        if (Cb) Cb[(size_t)row * N + col] = (__bf16)v;
        part += v * v;
      }
      if (rowsq) atomicAdd(rowsq + row, part);
    }
  }
}

// ---------------- per-text attribute Gram (10x10) ----------------
__global__ __launch_bounds__(256) void att_gram_kernel(const float* __restrict__ att, float* __restrict__ G) {
  __shared__ float red[256];
  int t = blockIdx.x, tid = threadIdx.x;
  const float* base = att + (size_t)t * 10 * 1024;
  for (int a = 0; a < 10; ++a)
    for (int b = a; b < 10; ++b) {
      float part = 0.f;
      for (int j = tid; j < 1024; j += 256) part += base[a * 1024 + j] * base[b * 1024 + j];
      red[tid] = part;
      __syncthreads();
      for (int off = 128; off > 0; off >>= 1) {
        if (tid < off) red[tid] += red[tid + off];
        __syncthreads();
      }
      if (tid == 0) { G[t * 100 + a * 10 + b] = red[0]; G[t * 100 + b * 10 + a] = red[0]; }
      __syncthreads();
    }
}

// ---------------- i2t: out[t,i] = dot_ti/10 (Gram trick, VALU) ----------------
__global__ __launch_bounds__(256) void i2t_kernel(
    const float* __restrict__ vis, const float* __restrict__ inv_vis,
    const float* __restrict__ att, const float* __restrict__ inv_att,
    const float* __restrict__ G, const int* __restrict__ att_nums, float* __restrict__ out) {
  __shared__ __align__(16) float sAtt[10 * 1024];
  __shared__ float sG[100];
  __shared__ float sInvA[10];
  int t = blockIdx.x, iblk = blockIdx.y;
  int tid = threadIdx.x, lane = tid & 31, wid = tid >> 5;
  for (int c = tid; c < 2560; c += 256) {
#if DH_ASYNC
    async_g2l_b128(att + (size_t)t * 10240 + c * 4, sAtt + c * 4);
#else
    *(float4*)(sAtt + c * 4) = *(const float4*)(att + (size_t)t * 10240 + c * 4);
#endif
  }
  if (tid < 100) sG[tid] = G[t * 100 + tid];
  if (tid < 10) sInvA[tid] = inv_att[t * 10 + tid];
  int an = att_nums[t];
  async_wait0();
  __syncthreads();
  for (int ii = 0; ii < 4; ++ii) {
    int i = iblk * 32 + ii * 8 + wid;
    float raw[10];
#pragma unroll
    for (int a = 0; a < 10; ++a) raw[a] = 0.f;
    for (int j = lane; j < 1024; j += 32) {
      float v = vis[(size_t)i * 1024 + j];
#pragma unroll
      for (int a = 0; a < 10; ++a) raw[a] += v * sAtt[a * 1024 + j];
    }
#pragma unroll
    for (int a = 0; a < 10; ++a)
#pragma unroll
      for (int off = 16; off > 0; off >>= 1) raw[a] += __shfl_xor(raw[a], off, 32);
    if (lane == 0) {
      float iv = inv_vis[i];
      float s[10]; float mx = -1e30f;
#pragma unroll
      for (int a = 0; a < 10; ++a) {
        float sa = (a < an) ? fmaxf(raw[a] * iv * sInvA[a], 0.f) * 20.f : -1e9f;
        s[a] = sa; mx = fmaxf(mx, sa);
      }
      float sum = 0.f;
#pragma unroll
      for (int a = 0; a < 10; ++a) { s[a] = __expf(s[a] - mx); sum += s[a]; }
      float rs = 1.f / sum;
      float num = 0.f;
#pragma unroll
      for (int a = 0; a < 10; ++a) { s[a] *= rs; num += s[a] * raw[a] * iv; }
      float den2 = 0.f;
#pragma unroll
      for (int a = 0; a < 10; ++a)
#pragma unroll
        for (int b = 0; b < 10; ++b) den2 += s[a] * s[b] * sG[a * 10 + b];
      float dot = num / fmaxf(sqrtf(fmaxf(den2, 0.f)), 1e-12f);
      out[t * 256 + i] = dot * 0.1f;
    }
  }
}

// ---------------- t2i: flash-style per image; out[t,i] += num/||P.patch_i|| ----------------
// LDS layout (dynamic):
//  [0,5120)        sA stage 64x40 bf16   (phase1)   | [0,16384) sO 64x64 f32 (phase3)
//  [5120,23040)    sB stage 224x40 bf16  (phase1)
//  [23040,80384)   sS 64x224 f32  (W values)
//  [80384,109056)  sP 64x224 bf16 (softmax probs)
//  [109056,137728) sPT 64x224 bf16 (patch chunk, [d][p])
//  [137728,138624) sInv 224 f32 ; [138624,138880) sNum ; [138880,139136) sDen
#define T2I_SMEM 139264
__global__ __launch_bounds__(256) void t2i_kernel(
    const __bf16* __restrict__ ntext, const __bf16* __restrict__ patchb,
    const float* __restrict__ invp, float* __restrict__ out) {
  extern __shared__ char smem[];
  __bf16* sA   = (__bf16*)(smem + 0);
  __bf16* sB   = (__bf16*)(smem + 5120);
  float*  sO   = (float*)(smem + 0);
  float*  sS   = (float*)(smem + 23040);
  __bf16* sP   = (__bf16*)(smem + 80384);
  __bf16* sPT  = (__bf16*)(smem + 109056);
  float*  sInv = (float*)(smem + 137728);
  float*  sNum = (float*)(smem + 138624);
  float*  sDen = (float*)(smem + 138880);

  const int i = blockIdx.x;
  const int t0 = blockIdx.y * 64;
  const int tid = threadIdx.x, lane = tid & 31, wid = tid >> 5;
  const int lh = lane >> 4, c16 = lane & 15;

  if (tid < 64) sDen[tid] = 0.f;
  for (int c = tid; c < 224; c += 256) sInv[c] = (c < 196) ? invp[i * 196 + c] : 0.f;

  // Phase 1: S[64][196] = ntext_tile x patch_i^T  (K = 1024)
  const int wm = wid >> 1, wn = wid & 1;
  v8f accS[7];
#pragma unroll
  for (int j = 0; j < 7; ++j) accS[j] = v8f_zero();

  for (int k0 = 0; k0 < 1024; k0 += 32) {
    __syncthreads();
    {
      int r = tid >> 2, kk = (tid & 3) * 8;  // 256 chunks exactly
      async_g2l_b128(ntext + (size_t)(t0 + r) * 1024 + k0 + kk, sA + r * 40 + kk);
    }
    for (int c = tid; c < 896; c += 256) {
      int p = c >> 2, kk = (c & 3) * 8;
      if (p < 196) async_g2l_b128(patchb + ((size_t)i * 196 + p) * 1024 + k0 + kk, sB + p * 40 + kk);
      else         *(v8bf*)(sB + p * 40 + kk) = v8bf_zero();
    }
    async_wait0();
    __syncthreads();
    v16bf a = frag_a(sA, 40, wm * 16, 0, lane);
#pragma unroll
    for (int j = 0; j < 7; ++j) {
      v16bf b = frag_b(sB, 40, (wn * 7 + j) * 16, 0, lane);
      accS[j] = wmma_bf16(a, b, accS[j]);
    }
  }
  __syncthreads();
#pragma unroll
  for (int j = 0; j < 7; ++j)
#pragma unroll
    for (int r = 0; r < 8; ++r)
      sS[(wm * 16 + lh * 8 + r) * 224 + (wn * 7 + j) * 16 + c16] = accS[j][r];
  __syncthreads();

  // Phase 2: per-row softmax over 196 patches + numerator
  if (tid < 64) {
    const float* Sr = sS + tid * 224;
    float mx = -1e30f;
    for (int p = 0; p < 196; ++p) mx = fmaxf(mx, fmaxf(Sr[p] * sInv[p], 0.f) * 20.f);
    float sum = 0.f;
    for (int p = 0; p < 196; ++p) sum += __expf(fmaxf(Sr[p] * sInv[p], 0.f) * 20.f - mx);
    float rs = 1.f / sum;
    float num = 0.f;
    for (int p = 0; p < 224; ++p) {
      float e = 0.f;
      if (p < 196) {
        e = __expf(fmaxf(Sr[p] * sInv[p], 0.f) * 20.f - mx) * rs;
        num += e * Sr[p];
      }
      sP[tid * 224 + p] = (__bf16)e;
    }
    sNum[tid] = num;
  }
  __syncthreads();

  // Phase 3: den = || P[64][196] x patch_i[196][1024] ||^2 over d-chunks of 64
  const int wm2 = wid >> 1, wn2 = wid & 1;
  for (int d0 = 0; d0 < 1024; d0 += 64) {
    for (int c = tid; c < 196 * 8; c += 256) {
      int p = c >> 3, q = c & 7;
      v8bf v = *(const v8bf*)(patchb + ((size_t)i * 196 + p) * 1024 + d0 + q * 8);
#pragma unroll
      for (int j = 0; j < 8; ++j) sPT[(q * 8 + j) * 224 + p] = v[j];
    }
    for (int c = tid; c < 64 * 28; c += 256) {
      int dd = c / 28, p = 196 + c % 28;
      sPT[dd * 224 + p] = (__bf16)0.f;
    }
    __syncthreads();
    v8f accO[2];
    accO[0] = v8f_zero(); accO[1] = v8f_zero();
    for (int ks = 0; ks < 224; ks += 32) {
      v16bf a = frag_a(sP, 224, wm2 * 16, ks, lane);
#pragma unroll
      for (int nt = 0; nt < 2; ++nt) {
        v16bf b = frag_b(sPT, 224, wn2 * 32 + nt * 16, ks, lane);
        accO[nt] = wmma_bf16(a, b, accO[nt]);
      }
    }
#pragma unroll
    for (int nt = 0; nt < 2; ++nt)
#pragma unroll
      for (int r = 0; r < 8; ++r)
        sO[(wm2 * 16 + lh * 8 + r) * 64 + wn2 * 32 + nt * 16 + c16] = accO[nt][r];
    __syncthreads();
    {
      int r = tid >> 2, q = tid & 3;
      float part = 0.f;
#pragma unroll
      for (int j = 0; j < 16; ++j) { float v = sO[r * 64 + q * 16 + j]; part += v * v; }
      atomicAdd(&sDen[r], part);
    }
    __syncthreads();
  }
  if (tid < 64) {
    float den = fmaxf(sqrtf(fmaxf(sDen[tid], 0.f)), 1e-12f);
    out[(t0 + tid) * 256 + i] += sNum[tid] / den;
  }
}

extern "C" void kernel_launch(void* const* d_in, const int* in_sizes, int n_in,
                              void* d_out, int out_size, void* d_ws, size_t ws_size,
                              hipStream_t stream) {
  (void)in_sizes; (void)n_in; (void)out_size; (void)ws_size;
  const float* visual  = (const float*)d_in[0];
  const float* textual = (const float*)d_in[1];
  const float* attrib  = (const float*)d_in[2];
  const int*   attn    = (const int*)d_in[3];
  const float* Wv = (const float*)d_in[4];  const float* bv = (const float*)d_in[5];
  const float* Wp = (const float*)d_in[6];  const float* bp = (const float*)d_in[7];
  const float* Wt = (const float*)d_in[8];  const float* bt = (const float*)d_in[9];
  const float* Wa = (const float*)d_in[10]; const float* ba = (const float*)d_in[11];
  float* out = (float*)d_out;

  char* w = (char*)d_ws;
  auto take = [&](size_t b) { char* p = w; w += (b + 255) & ~(size_t)255; return p; };
  __bf16* clsb   = (__bf16*)take((size_t)256 * 768 * 2);
  __bf16* pinb   = (__bf16*)take((size_t)50176 * 768 * 2);
  __bf16* textb  = (__bf16*)take((size_t)256 * 768 * 2);
  __bf16* attb   = (__bf16*)take((size_t)2560 * 768 * 2);
  __bf16* Wvb    = (__bf16*)take((size_t)1024 * 768 * 2);
  __bf16* Wpb    = (__bf16*)take((size_t)1024 * 768 * 2);
  __bf16* Wtb    = (__bf16*)take((size_t)1024 * 768 * 2);
  __bf16* Wab    = (__bf16*)take((size_t)1024 * 768 * 2);
  float*  visf   = (float*)take((size_t)256 * 1024 * 4);
  float*  textf  = (float*)take((size_t)256 * 1024 * 4);
  float*  attf   = (float*)take((size_t)2560 * 1024 * 4);
  __bf16* patchb = (__bf16*)take((size_t)50176 * 1024 * 2);
  __bf16* ntextb = (__bf16*)take((size_t)256 * 1024 * 2);
  float* sq_p  = (float*)take(50176 * 4);
  float* sq_v  = (float*)take(256 * 4);
  float* sq_t  = (float*)take(256 * 4);
  float* sq_a  = (float*)take(2560 * 4);
  float* inv_p = (float*)take(50176 * 4);
  float* inv_v = (float*)take(256 * 4);
  float* inv_t = (float*)take(256 * 4);
  float* inv_a = (float*)take(2560 * 4);
  float* G     = (float*)take(256 * 100 * 4);

  auto gs = [](int n) { return (n + 255) / 256; };

  // convert inputs to bf16
  split_visual_kernel<<<gs(256 * 197 * 768), 256, 0, stream>>>(visual, clsb, pinb, 256 * 197 * 768);
  f32_to_bf16_kernel<<<gs(256 * 768), 256, 0, stream>>>(textual, textb, 256 * 768);
  f32_to_bf16_kernel<<<gs(2560 * 768), 256, 0, stream>>>(attrib, attb, 2560 * 768);
  f32_to_bf16_kernel<<<gs(1024 * 768), 256, 0, stream>>>(Wv, Wvb, 1024 * 768);
  f32_to_bf16_kernel<<<gs(1024 * 768), 256, 0, stream>>>(Wp, Wpb, 1024 * 768);
  f32_to_bf16_kernel<<<gs(1024 * 768), 256, 0, stream>>>(Wt, Wtb, 1024 * 768);
  f32_to_bf16_kernel<<<gs(1024 * 768), 256, 0, stream>>>(Wa, Wab, 1024 * 768);

  zero_f32_kernel<<<gs(50176), 256, 0, stream>>>(sq_p, 50176);
  zero_f32_kernel<<<gs(256), 256, 0, stream>>>(sq_v, 256);
  zero_f32_kernel<<<gs(256), 256, 0, stream>>>(sq_t, 256);
  zero_f32_kernel<<<gs(2560), 256, 0, stream>>>(sq_a, 2560);

  // embedding GEMMs (WMMA bf16, fp32 accum) + fused row sumsq
  gemm_bf16_kernel<<<dim3(2, 8),   256, 0, stream>>>(clsb,  Wvb, bv, visf,   nullptr, sq_v, 256,   1024, 768);
  gemm_bf16_kernel<<<dim3(392, 8), 256, 0, stream>>>(pinb,  Wpb, bp, nullptr, patchb, sq_p, 50176, 1024, 768);
  gemm_bf16_kernel<<<dim3(2, 8),   256, 0, stream>>>(textb, Wtb, bt, textf,  nullptr, sq_t, 256,   1024, 768);
  gemm_bf16_kernel<<<dim3(20, 8),  256, 0, stream>>>(attb,  Wab, ba, attf,   nullptr, sq_a, 2560,  1024, 768);

  inv_norm_kernel<<<gs(50176), 256, 0, stream>>>(sq_p, inv_p, 50176);
  inv_norm_kernel<<<gs(256), 256, 0, stream>>>(sq_v, inv_v, 256);
  inv_norm_kernel<<<gs(256), 256, 0, stream>>>(sq_t, inv_t, 256);
  inv_norm_kernel<<<gs(2560), 256, 0, stream>>>(sq_a, inv_a, 2560);

  scale_rows_bf16_kernel<<<gs(256 * 1024), 256, 0, stream>>>(textf, inv_t, ntextb, 256 * 1024, 1024);
  att_gram_kernel<<<256, 256, 0, stream>>>(attf, G);

  // i2t writes out (=), t2i accumulates (+=) — same stream, ordered.
  i2t_kernel<<<dim3(256, 8), 256, 0, stream>>>(visf, inv_v, attf, inv_a, G, attn, out);
  t2i_kernel<<<dim3(256, 4), 256, T2I_SMEM, stream>>>(ntextb, patchb, inv_p, out);
}
